// MARLNetwork_87162066305432
// MI455X (gfx1250) — matrix operand order repaired and checked by reading, not compiled, not run on hardware
//
#include <hip/hip_runtime.h>
#include <hip/hip_bf16.h>
#include <math.h>

typedef _Float16 half_t;
typedef _Float16 v16h __attribute__((ext_vector_type(16)));
typedef float    v8f  __attribute__((ext_vector_type(8)));

#define BATCH   8
#define NNODE   1024
#define BN      8192      // BATCH*NNODE
#define FEAT    1568
#define HID     64
#define NACT    16
#define KB_FEAT 49        // 1568/32
#define KB_ATT  32        // 1024/32
#define NEGBIG  (-9.0e15f)
#define ALPHA   0.2f

// ---------------------------------------------------------------------------
// Kernel 1: per-node CNN (conv1 3x3 SAME +relu, conv2 3x3 VALID +relu),
// writes features as f16 in WMMA A-fragment packed layout:
//   A 16x32 f16: lane = (m%16) + (K in {8..15,24..31} ? 16:0)
//                elem = (K&7) + (K>=16 ? 8:0)
// ---------------------------------------------------------------------------
__global__ __launch_bounds__(256)
void cnn_kernel(const float* __restrict__ obs,
                const float* __restrict__ w1, const float* __restrict__ b1,
                const float* __restrict__ w2, const float* __restrict__ b2,
                half_t* __restrict__ featsP) {
  __shared__ float s_w1[144];
  __shared__ float s_b1[16];
  __shared__ float s_w2[4608];
  __shared__ float s_b2[32];
  __shared__ float s_img[8][81];
  __shared__ float s_c1[8][1296];
  int tid = threadIdx.x;
  for (int i = tid; i < 144;  i += 256) s_w1[i] = w1[i];
  for (int i = tid; i < 16;   i += 256) s_b1[i] = b1[i];
  for (int i = tid; i < 4608; i += 256) s_w2[i] = w2[i];
  for (int i = tid; i < 32;   i += 256) s_b2[i] = b2[i];
  int wv = tid >> 5, lane = tid & 31;
  int m = blockIdx.x * 8 + wv;                      // image index 0..8191
  const float* src = obs + (size_t)m * 81;
  for (int i = lane; i < 81; i += 32) s_img[wv][i] = src[i];
  __syncthreads();
  // conv1: 16ch x 9x9, SAME padding
  for (int idx = lane; idx < 1296; idx += 32) {
    int ch = idx / 81, pos = idx % 81;
    int oy = pos / 9, ox = pos % 9;
    float acc = s_b1[ch];
    for (int ky = 0; ky < 3; ++ky) {
      int iy = oy + ky - 1;
      if (iy < 0 || iy > 8) continue;
      for (int kx = 0; kx < 3; ++kx) {
        int ix = ox + kx - 1;
        if (ix < 0 || ix > 8) continue;
        acc += s_img[wv][iy * 9 + ix] * s_w1[ch * 9 + ky * 3 + kx];
      }
    }
    s_c1[wv][idx] = acc > 0.f ? acc : 0.f;
  }
  __syncthreads();
  // conv2: 32ch x 7x7 VALID over 16 in-ch; pack output features
  int mt = m >> 4, mr = m & 15;
  for (int o = lane; o < FEAT; o += 32) {
    int ch = o / 49, pos = o % 49;
    int oy = pos / 7, ox = pos % 7;
    float acc = s_b2[ch];
    for (int ic = 0; ic < 16; ++ic) {
      const float* cp = &s_c1[wv][ic * 81 + oy * 9 + ox];
      const float* wp = &s_w2[(ch * 16 + ic) * 9];
      #pragma unroll
      for (int ky = 0; ky < 3; ++ky)
        #pragma unroll
        for (int kx = 0; kx < 3; ++kx)
          acc += cp[ky * 9 + kx] * wp[ky * 3 + kx];
    }
    acc = acc > 0.f ? acc : 0.f;
    int kb = o >> 5, kr = o & 31;
    int lt = mr + (((kr >> 3) & 1) ? 16 : 0);
    int el = (kr & 7) + ((kr >= 16) ? 8 : 0);
    featsP[(((size_t)mt * KB_FEAT + kb) * 32 + lt) * 16 + el] = (half_t)acc;
  }
}

// ---------------------------------------------------------------------------
// Kernel 2: pack W (1568x64) into f16 B-fragment layout:
//   B 32x16 f16: lane = (n%16) + (K in {8..15,24..31} ? 16:0), elem = (K&7) + (K>=16 ? 8:0)
// ---------------------------------------------------------------------------
__global__ void pack_w_kernel(const float* __restrict__ W, half_t* __restrict__ Wp) {
  int idx = blockIdx.x * blockDim.x + threadIdx.x;
  if (idx >= FEAT * HID) return;
  int k = idx / HID, n = idx % HID;
  int kb = k >> 5, kr = k & 31;
  int nt = n >> 4, nr = n & 15;
  int lane = nr + (((kr >> 3) & 1) ? 16 : 0);
  int el = (kr & 7) + ((kr >= 16) ? 8 : 0);
  Wp[(((size_t)nt * KB_FEAT + kb) * 32 + lane) * 16 + el] = (half_t)W[idx];
}

// ---------------------------------------------------------------------------
// Kernel 3: projection GEMM Wh = feats @ W  (8192x1568 @ 1568x64) via WMMA.
// Wave computes a 16x64 strip (4 accumulators). K-loop kept rolled so the
// accumulators stay pinned (WMMA D==C in place, no copy chains).
// Epilogue writes Wh (f32) and repacks Wh as B-fragments for attention.
// ---------------------------------------------------------------------------
__global__ __launch_bounds__(256)
void proj_gemm_kernel(const half_t* __restrict__ Ap, const half_t* __restrict__ Bp,
                      float* __restrict__ Wh, half_t* __restrict__ WhP) {
  int wv = threadIdx.x >> 5, lane = threadIdx.x & 31;
  int tile = blockIdx.x * 8 + wv;                   // 512 tiles of 16 rows
  v8f acc0, acc1, acc2, acc3;
  #pragma unroll
  for (int q = 0; q < 8; ++q) { acc0[q] = 0.f; acc1[q] = 0.f; acc2[q] = 0.f; acc3[q] = 0.f; }
  const v16h* aBase = (const v16h*)Ap + (size_t)tile * KB_FEAT * 32 + lane;
  const v16h* bBase = (const v16h*)Bp + lane;
  #pragma unroll 1
  for (int kb = 0; kb < KB_FEAT; ++kb) {
    v16h a  = aBase[(size_t)kb * 32];
    v16h b0 = bBase[(size_t)(0 * KB_FEAT + kb) * 32];
    v16h b1 = bBase[(size_t)(1 * KB_FEAT + kb) * 32];
    v16h b2 = bBase[(size_t)(2 * KB_FEAT + kb) * 32];
    v16h b3 = bBase[(size_t)(3 * KB_FEAT + kb) * 32];
    acc0 = __builtin_amdgcn_wmma_f32_16x16x32_f16(false, a, false, b0, (short)0, acc0, false, false);
    acc1 = __builtin_amdgcn_wmma_f32_16x16x32_f16(false, a, false, b1, (short)0, acc1, false, false);
    acc2 = __builtin_amdgcn_wmma_f32_16x16x32_f16(false, a, false, b2, (short)0, acc2, false, false);
    acc3 = __builtin_amdgcn_wmma_f32_16x16x32_f16(false, a, false, b3, (short)0, acc3, false, false);
  }
  int halfsel = (lane >= 16) ? 8 : 0;
  int nl = lane & 15;
  v8f accs[4] = {acc0, acc1, acc2, acc3};
  #pragma unroll
  for (int nt = 0; nt < 4; ++nt) {
    int n = nt * 16 + nl;
    #pragma unroll
    for (int r = 0; r < 8; ++r) {
      int mrow = tile * 16 + r + halfsel;
      float v = accs[nt][r];
      Wh[(size_t)mrow * HID + n] = v;
      int bb = mrow >> 10, j = mrow & 1023;
      int kb2 = j >> 5, kr = j & 31;
      int lane2 = nl + (((kr >> 3) & 1) ? 16 : 0);
      int el = (kr & 7) + ((kr >= 16) ? 8 : 0);
      WhP[((((size_t)bb * 4 + nt) * KB_ATT + kb2) * 32 + lane2) * 16 + el] = (half_t)v;
    }
  }
}

// ---------------------------------------------------------------------------
// Kernel 4: s1 = Wh@a1, s2 = Wh@a2 (per-node scalars)
// ---------------------------------------------------------------------------
__global__ void s12_kernel(const float* __restrict__ Wh, const float* __restrict__ a1,
                           const float* __restrict__ a2,
                           float* __restrict__ s1, float* __restrict__ s2) {
  int m = blockIdx.x * blockDim.x + threadIdx.x;
  if (m >= BN) return;
  const float* row = Wh + (size_t)m * HID;
  float r1 = 0.f, r2 = 0.f;
  #pragma unroll 8
  for (int d = 0; d < HID; ++d) { float v = row[d]; r1 += v * a1[d]; r2 += v * a2[d]; }
  s1[m] = r1; s2[m] = r2;
}

// ---------------------------------------------------------------------------
// Kernel 5: per-row softmax stats (max, 1/Z) with exact masked semantics:
//   e = adj>0 ? leakyrelu(s1_i + s2_j) : NEG   (so all-masked row -> uniform)
// One wave per row.
// ---------------------------------------------------------------------------
__global__ __launch_bounds__(256)
void softmax_stats_kernel(const float* __restrict__ s1, const float* __restrict__ s2,
                          const int* __restrict__ adj,
                          float* __restrict__ rowMax, float* __restrict__ rowInvZ) {
  __shared__ float s2sh[NNODE];
  int wv = threadIdx.x >> 5, lane = threadIdx.x & 31;
  int row = blockIdx.x * 8 + wv;                    // 0..8191
  int b = row >> 10;
  for (int i = threadIdx.x; i < NNODE; i += 256) s2sh[i] = s2[b * NNODE + i];
  __syncthreads();
  float s1v = s1[row];
  const int* arow = adj + (size_t)row * NNODE;
  float evals[32];
  float mx = -3.4e38f;
  #pragma unroll
  for (int k = 0; k < 32; ++k) {
    int j = k * 32 + lane;
    float x = s1v + s2sh[j];
    float e = x > 0.f ? x : ALPHA * x;
    e = (arow[j] > 0) ? e : NEGBIG;
    evals[k] = e;
    mx = fmaxf(mx, e);
  }
  #pragma unroll
  for (int off = 16; off > 0; off >>= 1) mx = fmaxf(mx, __shfl_xor(mx, off, 32));
  float z = 0.f;
  #pragma unroll
  for (int k = 0; k < 32; ++k) z += __expf(evals[k] - mx);
  #pragma unroll
  for (int off = 16; off > 0; off >>= 1) z += __shfl_xor(z, off, 32);
  if (lane == 0) { rowMax[row] = mx; rowInvZ[row] = 1.f / z; }
}

// ---------------------------------------------------------------------------
// Kernel 6: attention GEMM  emb = elu( softmax(e) @ Wh ) via WMMA.
// Wave per 16-row tile; builds probability A-fragments on the fly (f16),
// accumulates against packed Wh B-fragments over 32 K-blocks, then scales
// by 1/Z (softmax normalization distributes over the GEMM) and applies ELU.
// ---------------------------------------------------------------------------
__global__ __launch_bounds__(256)
void attn_kernel(const float* __restrict__ s1, const float* __restrict__ s2,
                 const int* __restrict__ adj, const half_t* __restrict__ WhP,
                 const float* __restrict__ rowMax, const float* __restrict__ rowInvZ,
                 float* __restrict__ emb) {
  __shared__ float s2sh[NNODE];
  int wv = threadIdx.x >> 5, lane = threadIdx.x & 31;
  int tile = blockIdx.x * 8 + wv;                   // 512 tiles
  int rowBase = tile * 16;
  int b = rowBase >> 10;
  for (int i = threadIdx.x; i < NNODE; i += 256) s2sh[i] = s2[b * NNODE + i];
  __syncthreads();
  int mr = lane & 15;
  int i = rowBase + mr;                             // A-fragment row of this lane
  float s1v = s1[i];
  float mxv = rowMax[i];
  int hi8 = (lane >= 16) ? 8 : 0;
  const int* arow = adj + ((size_t)b * NNODE + (i & 1023)) * NNODE;
  v8f acc0, acc1, acc2, acc3;
  #pragma unroll
  for (int q = 0; q < 8; ++q) { acc0[q] = 0.f; acc1[q] = 0.f; acc2[q] = 0.f; acc3[q] = 0.f; }
  const v16h* bBase = (const v16h*)WhP + (size_t)b * 4 * KB_ATT * 32 + lane;
  #pragma unroll 1
  for (int kb = 0; kb < KB_ATT; ++kb) {
    int j0 = kb * 32 + hi8;
    int4 q0 = *(const int4*)(arow + j0);
    int4 q1 = *(const int4*)(arow + j0 + 4);
    int4 q2 = *(const int4*)(arow + j0 + 16);
    int4 q3 = *(const int4*)(arow + j0 + 20);
    int adjv[16] = {q0.x, q0.y, q0.z, q0.w, q1.x, q1.y, q1.z, q1.w,
                    q2.x, q2.y, q2.z, q2.w, q3.x, q3.y, q3.z, q3.w};
    v16h a;
    #pragma unroll
    for (int e = 0; e < 16; ++e) {
      int j = j0 + ((e < 8) ? e : (16 + e - 8));    // A-frag K mapping
      float x = s1v + s2sh[j];
      float ev = x > 0.f ? x : ALPHA * x;
      ev = (adjv[e] > 0) ? ev : NEGBIG;
      a[e] = (half_t)__expf(ev - mxv);              // <=1, underflow->0 for masked
    }
    v16h b0 = bBase[(size_t)(0 * KB_ATT + kb) * 32];
    v16h b1 = bBase[(size_t)(1 * KB_ATT + kb) * 32];
    v16h b2 = bBase[(size_t)(2 * KB_ATT + kb) * 32];
    v16h b3 = bBase[(size_t)(3 * KB_ATT + kb) * 32];
    acc0 = __builtin_amdgcn_wmma_f32_16x16x32_f16(false, a, false, b0, (short)0, acc0, false, false);
    acc1 = __builtin_amdgcn_wmma_f32_16x16x32_f16(false, a, false, b1, (short)0, acc1, false, false);
    acc2 = __builtin_amdgcn_wmma_f32_16x16x32_f16(false, a, false, b2, (short)0, acc2, false, false);
    acc3 = __builtin_amdgcn_wmma_f32_16x16x32_f16(false, a, false, b3, (short)0, acc3, false, false);
  }
  int halfsel = (lane >= 16) ? 8 : 0;
  v8f accs[4] = {acc0, acc1, acc2, acc3};
  #pragma unroll
  for (int r = 0; r < 8; ++r) {
    int mrow = rowBase + r + halfsel;
    float inv = rowInvZ[mrow];
    #pragma unroll
    for (int nt = 0; nt < 4; ++nt) {
      float v = accs[nt][r] * inv;
      v = v > 0.f ? v : (__expf(v) - 1.f);          // ELU
      emb[(size_t)mrow * HID + nt * 16 + mr] = v;
    }
  }
}

// ---------------------------------------------------------------------------
// Kernel 7: heads. logits = emb@actor_w + ab ; values = emb@critic_w + cb
// d_out = [logits (8*1024*16) | values (8*1024)]
// ---------------------------------------------------------------------------
__global__ __launch_bounds__(256)
void heads_kernel(const float* __restrict__ emb, const float* __restrict__ aw,
                  const float* __restrict__ ab, const float* __restrict__ cw,
                  const float* __restrict__ cb, float* __restrict__ out) {
  __shared__ float s_aw[HID * NACT];
  __shared__ float s_ab[NACT];
  __shared__ float s_cw[HID];
  for (int idx = threadIdx.x; idx < HID * NACT; idx += 256) s_aw[idx] = aw[idx];
  if (threadIdx.x < NACT) s_ab[threadIdx.x] = ab[threadIdx.x];
  for (int idx = threadIdx.x; idx < HID; idx += 256) s_cw[idx] = cw[idx];
  __syncthreads();
  int m = blockIdx.x * 256 + threadIdx.x;
  if (m >= BN) return;
  const float* row = emb + (size_t)m * HID;
  float e[HID];
  #pragma unroll
  for (int d = 0; d < HID; ++d) e[d] = row[d];
  float* logits = out + (size_t)m * NACT;
  for (int a = 0; a < NACT; ++a) {
    float acc = s_ab[a];
    #pragma unroll
    for (int d = 0; d < HID; ++d) acc += e[d] * s_aw[d * NACT + a];
    logits[a] = acc;
  }
  float v = cb[0];
  #pragma unroll
  for (int d = 0; d < HID; ++d) v += e[d] * s_cw[d];
  out[(size_t)BN * NACT + m] = v;
}

// ---------------------------------------------------------------------------
extern "C" void kernel_launch(void* const* d_in, const int* in_sizes, int n_in,
                              void* d_out, int out_size, void* d_ws, size_t ws_size,
                              hipStream_t stream) {
  const float* obs = (const float*)d_in[0];
  const int*   adj = (const int*)d_in[1];
  const float* c1w = (const float*)d_in[2];
  const float* c1b = (const float*)d_in[3];
  const float* c2w = (const float*)d_in[4];
  const float* c2b = (const float*)d_in[5];
  const float* W   = (const float*)d_in[6];
  const float* a1  = (const float*)d_in[7];
  const float* a2  = (const float*)d_in[8];
  const float* aw  = (const float*)d_in[9];
  const float* ab  = (const float*)d_in[10];
  const float* cw  = (const float*)d_in[11];
  const float* cb  = (const float*)d_in[12];

  char* ws = (char*)d_ws;
  auto alloc = [&](size_t bytes) {
    char* p = ws;
    ws += (bytes + 255) & ~(size_t)255;
    return p;
  };
  half_t* featsP = (half_t*)alloc((size_t)BN * FEAT * sizeof(half_t));   // packed A
  half_t* Wp     = (half_t*)alloc((size_t)FEAT * HID * sizeof(half_t));  // packed B
  float*  Wh     = (float*) alloc((size_t)BN * HID * sizeof(float));
  half_t* WhP    = (half_t*)alloc((size_t)BN * HID * sizeof(half_t));    // packed B (attn)
  float*  s1     = (float*) alloc((size_t)BN * sizeof(float));
  float*  s2     = (float*) alloc((size_t)BN * sizeof(float));
  float*  rmax   = (float*) alloc((size_t)BN * sizeof(float));
  float*  rinv   = (float*) alloc((size_t)BN * sizeof(float));
  float*  emb    = (float*) alloc((size_t)BN * HID * sizeof(float));

  hipLaunchKernelGGL(cnn_kernel, dim3(1024), dim3(256), 0, stream,
                     obs, c1w, c1b, c2w, c2b, featsP);
  hipLaunchKernelGGL(pack_w_kernel, dim3((FEAT * HID + 255) / 256), dim3(256), 0, stream,
                     W, Wp);
  hipLaunchKernelGGL(proj_gemm_kernel, dim3(64), dim3(256), 0, stream,
                     featsP, Wp, Wh, WhP);
  hipLaunchKernelGGL(s12_kernel, dim3(BN / 256), dim3(256), 0, stream,
                     Wh, a1, a2, s1, s2);
  hipLaunchKernelGGL(softmax_stats_kernel, dim3(1024), dim3(256), 0, stream,
                     s1, s2, adj, rmax, rinv);
  hipLaunchKernelGGL(attn_kernel, dim3(64), dim3(256), 0, stream,
                     s1, s2, adj, WhP, rmax, rinv, emb);
  hipLaunchKernelGGL(heads_kernel, dim3(BN / 256), dim3(256), 0, stream,
                     emb, aw, ab, cw, cb, (float*)d_out);
}
